// CausalSelfAttention_57801669870006
// MI455X (gfx1250) — matrix operand order repaired and checked by reading, not compiled, not run on hardware
//
#include <hip/hip_runtime.h>

// ---------------------------------------------------------------------------
// Types matching gfx1250 WMMA builtin signatures
// ---------------------------------------------------------------------------
typedef __bf16 v8bf  __attribute__((ext_vector_type(8)));
typedef __bf16 v16bf __attribute__((ext_vector_type(16)));
typedef float  v8f   __attribute__((ext_vector_type(8)));

__device__ __forceinline__ v8f wmma_bf16(v16bf a, v16bf b, v8f c) {
    // D = A(16x32 bf16) * B(32x16 bf16) + C(16x16 f32)
    return __builtin_amdgcn_wmma_f32_16x16x32_bf16(
        /*neg_a=*/false, a, /*neg_b=*/false, b,
        /*c_mod=*/(short)0, c, /*reuse_a=*/false, /*reuse_b=*/false);
}

__device__ __forceinline__ unsigned short f32_to_bf16(float f) {
    union { float f; unsigned int u; } x; x.f = f;
    unsigned int r = x.u + 0x7FFFu + ((x.u >> 16) & 1u);   // RNE
    return (unsigned short)(r >> 16);
}

// A-fragment (16-bit 16x32): lane holds row M = lane%16.
// lanes 0-15: K = [0..16);  lanes 16-31: K = [16..32)
__device__ __forceinline__ v16bf load_a_frag(const unsigned short* rowk, int hi) {
    v8bf lo = *(const v8bf*)(rowk + hi * 16);
    v8bf hs = *(const v8bf*)(rowk + hi * 16 + 8);
    return __builtin_shufflevector(lo, hs, 0,1,2,3,4,5,6,7,8,9,10,11,12,13,14,15);
}

__device__ __forceinline__ float hmax16(float x) {
    x = fmaxf(x, __shfl_xor(x, 1, 32));
    x = fmaxf(x, __shfl_xor(x, 2, 32));
    x = fmaxf(x, __shfl_xor(x, 4, 32));
    x = fmaxf(x, __shfl_xor(x, 8, 32));
    return x;
}
__device__ __forceinline__ float hsum16(float x) {
    x += __shfl_xor(x, 1, 32);
    x += __shfl_xor(x, 2, 32);
    x += __shfl_xor(x, 4, 32);
    x += __shfl_xor(x, 8, 32);
    return x;
}

// ---------------------------------------------------------------------------
// Problem constants
// ---------------------------------------------------------------------------
#define TOKENS 4096   // B*T
#define CMODEL 1024
#define NQKV   3072
#define SEQ    2048
#define DHEAD  64
#define NHEAD  16

// ---------------------------------------------------------------------------
// Kernel 0: f32 -> bf16 convert
// ---------------------------------------------------------------------------
__global__ __launch_bounds__(256) void cvt_bf16_kernel(const float* __restrict__ src,
                                                       unsigned short* __restrict__ dst,
                                                       int n) {
    int i = blockIdx.x * 256 + threadIdx.x;
    if (i < n) dst[i] = f32_to_bf16(src[i]);
}

// ---------------------------------------------------------------------------
// Kernel 1: QKV projection. C[m,n] = sum_k x[m,k] * w_qkv[n,k]
// One wave -> 32x64 output tile: 2 A-fragments x 4 B-fragments -> 8 WMMAs/K-step
// Epilogue scatters into Q [B,H,T,64] (pre-scaled), K [B,H,T,64], Vt [B,H,64,T]
// ---------------------------------------------------------------------------
__global__ __launch_bounds__(256) void qkv_gemm_kernel(
    const unsigned short* __restrict__ xb, const unsigned short* __restrict__ wb,
    unsigned short* __restrict__ Qb, unsigned short* __restrict__ Kb,
    unsigned short* __restrict__ Vtb) {
    int wave = blockIdx.x * 8 + (threadIdx.x >> 5);
    int lane = threadIdx.x & 31;
    int hi = lane >> 4, n = lane & 15;
    int mT = wave / (NQKV / 64);          // 128 row tiles of 32
    int nT = wave % (NQKV / 64);          // 48 col tiles of 64

    const unsigned short* arow0 = xb + (size_t)(mT * 32 + n) * CMODEL;
    const unsigned short* arow1 = arow0 + (size_t)16 * CMODEL;
    const unsigned short* brow  = wb + (size_t)(nT * 64 + n) * CMODEL + hi * 16;

    v8f acc[2][4];
#pragma unroll
    for (int r = 0; r < 2; ++r)
#pragma unroll
        for (int c = 0; c < 4; ++c) acc[r][c] = (v8f){};

    for (int k = 0; k < CMODEL; k += 32) {
        v16bf a0 = load_a_frag(arow0 + k, hi);
        v16bf a1 = load_a_frag(arow1 + k, hi);
        v16bf b0 = *(const v16bf*)(brow + k);
        v16bf b1 = *(const v16bf*)(brow + 16 * CMODEL + k);
        v16bf b2 = *(const v16bf*)(brow + 32 * CMODEL + k);
        v16bf b3 = *(const v16bf*)(brow + 48 * CMODEL + k);
        acc[0][0] = wmma_bf16(a0, b0, acc[0][0]);
        acc[1][0] = wmma_bf16(a1, b0, acc[1][0]);
        acc[0][1] = wmma_bf16(a0, b1, acc[0][1]);
        acc[1][1] = wmma_bf16(a1, b1, acc[1][1]);
        acc[0][2] = wmma_bf16(a0, b2, acc[0][2]);
        acc[1][2] = wmma_bf16(a1, b2, acc[1][2]);
        acc[0][3] = wmma_bf16(a0, b3, acc[0][3]);
        acc[1][3] = wmma_bf16(a1, b3, acc[1][3]);
    }

#pragma unroll
    for (int r = 0; r < 2; ++r) {
#pragma unroll
        for (int c = 0; c < 4; ++c) {
            int col  = nT * 64 + c * 16 + n;
            int sect = col >> 10;           // 0=Q 1=K 2=V
            int col1 = col & 1023;
            int h = col1 >> 6, dd = col1 & 63;
#pragma unroll
            for (int v = 0; v < 8; ++v) {
                int t  = mT * 32 + r * 16 + v + 8 * hi;   // global token
                int bi = t >> 11, tt = t & 2047;
                int bh = bi * NHEAD + h;
                float val = acc[r][c][v];
                if (sect == 0)
                    Qb[((size_t)bh * SEQ + tt) * DHEAD + dd] = f32_to_bf16(val * 0.125f);
                else if (sect == 1)
                    Kb[((size_t)bh * SEQ + tt) * DHEAD + dd] = f32_to_bf16(val);
                else
                    Vtb[((size_t)bh * DHEAD + dd) * SEQ + tt] = f32_to_bf16(val);
            }
        }
    }
}

// ---------------------------------------------------------------------------
// Kernel 2: flash attention. One wave per (batch*head, 16-query block).
// ---------------------------------------------------------------------------
__global__ __launch_bounds__(256) void attn_kernel(
    const unsigned short* __restrict__ Qb, const unsigned short* __restrict__ Kb,
    const unsigned short* __restrict__ Vtb, unsigned short* __restrict__ attnb) {
    __shared__ unsigned short plds[8 * 16 * 32];   // per-wave 16x32 bf16 P tile

    int wslot = threadIdx.x >> 5;
    int wave  = blockIdx.x * 8 + wslot;
    int lane  = threadIdx.x & 31;
    int hi = lane >> 4, n = lane & 15;
    int qb = wave & (SEQ / 16 - 1);      // query block
    int bh = wave >> 7;                  // batch*head

    const unsigned short* Q  = Qb  + (size_t)bh * SEQ * DHEAD;
    const unsigned short* Km = Kb  + (size_t)bh * SEQ * DHEAD;
    const unsigned short* Vt = Vtb + (size_t)bh * DHEAD * SEQ;
    unsigned short* pt = plds + wslot * 512;

    int qbase = qb * 16;
    const unsigned short* qrow = Q + (size_t)(qbase + n) * DHEAD;
    v16bf qa0 = load_a_frag(qrow, hi);        // d = 0..31
    v16bf qa1 = load_a_frag(qrow + 32, hi);   // d = 32..63

    v8f o0 = {}, o1 = {}, o2 = {}, o3 = {};
    float m[8], l[8];
#pragma unroll
    for (int v = 0; v < 8; ++v) { m[v] = -__builtin_inff(); l[v] = 0.0f; }

    int rbase = qbase + hi * 8;
    for (int kb = 0; kb < qbase + 16; kb += 32) {
        // ---- S = Q * K^T for keys [kb, kb+32) (Q already holds the 1/8 scale)
        v8f s0 = {}, s1 = {};
        {
            const unsigned short* kr = Km + (size_t)(kb + n) * DHEAD + hi * 16;
            v16bf b0 = *(const v16bf*)(kr);
            v16bf b1 = *(const v16bf*)(kr + 32);
            s0 = wmma_bf16(qa0, b0, s0);
            s0 = wmma_bf16(qa1, b1, s0);
            const unsigned short* kr2 = kr + 16 * DHEAD;
            v16bf c0 = *(const v16bf*)(kr2);
            v16bf c1 = *(const v16bf*)(kr2 + 32);
            s1 = wmma_bf16(qa0, c0, s1);
            s1 = wmma_bf16(qa1, c1, s1);
        }
        // ---- causal mask + online softmax (rows live in 16-lane halves)
#pragma unroll
        for (int v = 0; v < 8; ++v) {
            int r = rbase + v;
            if (kb + n      > r) s0[v] = -__builtin_inff();
            if (kb + 16 + n > r) s1[v] = -__builtin_inff();
        }
#pragma unroll
        for (int v = 0; v < 8; ++v) {
            float mx = hmax16(fmaxf(s0[v], s1[v]));
            float mn = fmaxf(m[v], mx);             // finite: key kb<=r always live
            float alpha = __expf(m[v] - mn);
            float p0 = __expf(s0[v] - mn);
            float p1 = __expf(s1[v] - mn);
            l[v] = l[v] * alpha + hsum16(p0 + p1);
            m[v] = mn;
            o0[v] *= alpha; o1[v] *= alpha; o2[v] *= alpha; o3[v] *= alpha;
            s0[v] = p0; s1[v] = p1;
        }
        // ---- stage P (C-layout) through LDS, reload as A-fragment
#pragma unroll
        for (int v = 0; v < 8; ++v) {
            int r = v + 8 * hi;
            pt[r * 32 + n]      = f32_to_bf16(s0[v]);
            pt[r * 32 + 16 + n] = f32_to_bf16(s1[v]);
        }
        asm volatile("s_wait_dscnt 0" ::: "memory");
        v16bf pa = load_a_frag(pt + n * 32, hi);
        asm volatile("" ::: "memory");
        // ---- O += P * V   (V transposed: B-fragments are contiguous runs)
        const unsigned short* vb = Vt + (size_t)n * SEQ + kb + hi * 16;
        o0 = wmma_bf16(pa, *(const v16bf*)(vb),            o0);
        o1 = wmma_bf16(pa, *(const v16bf*)(vb + 16 * SEQ), o1);
        o2 = wmma_bf16(pa, *(const v16bf*)(vb + 32 * SEQ), o2);
        o3 = wmma_bf16(pa, *(const v16bf*)(vb + 48 * SEQ), o3);
    }

    // ---- normalize and write merged-head layout [B*T, C] bf16
    int bi = bh >> 4, h = bh & 15;
#pragma unroll
    for (int v = 0; v < 8; ++v) {
        int r = qbase + v + 8 * hi;
        float inv = 1.0f / l[v];
        size_t off = ((size_t)(bi * SEQ + r)) * CMODEL + h * DHEAD + n;
        attnb[off]      = f32_to_bf16(o0[v] * inv);
        attnb[off + 16] = f32_to_bf16(o1[v] * inv);
        attnb[off + 32] = f32_to_bf16(o2[v] * inv);
        attnb[off + 48] = f32_to_bf16(o3[v] * inv);
    }
}

// ---------------------------------------------------------------------------
// Kernel 3: output projection. out[m,n] = sum_k attn[m,k] * w_out[n,k]  (f32 out)
// One wave -> 32x64 output tile, 8 WMMAs per K-step.
// ---------------------------------------------------------------------------
__global__ __launch_bounds__(256) void out_gemm_kernel(
    const unsigned short* __restrict__ ab, const unsigned short* __restrict__ wb,
    float* __restrict__ out) {
    int wave = blockIdx.x * 8 + (threadIdx.x >> 5);
    int lane = threadIdx.x & 31;
    int hi = lane >> 4, n = lane & 15;
    int mT = wave >> 4;        // 128 row tiles of 32
    int nT = wave & 15;        // 16 col tiles of 64

    const unsigned short* arow0 = ab + (size_t)(mT * 32 + n) * CMODEL;
    const unsigned short* arow1 = arow0 + (size_t)16 * CMODEL;
    const unsigned short* brow  = wb + (size_t)(nT * 64 + n) * CMODEL + hi * 16;

    v8f acc[2][4];
#pragma unroll
    for (int r = 0; r < 2; ++r)
#pragma unroll
        for (int c = 0; c < 4; ++c) acc[r][c] = (v8f){};

    for (int k = 0; k < CMODEL; k += 32) {
        v16bf a0 = load_a_frag(arow0 + k, hi);
        v16bf a1 = load_a_frag(arow1 + k, hi);
        v16bf b0 = *(const v16bf*)(brow + k);
        v16bf b1 = *(const v16bf*)(brow + 16 * CMODEL + k);
        v16bf b2 = *(const v16bf*)(brow + 32 * CMODEL + k);
        v16bf b3 = *(const v16bf*)(brow + 48 * CMODEL + k);
        acc[0][0] = wmma_bf16(a0, b0, acc[0][0]);
        acc[1][0] = wmma_bf16(a1, b0, acc[1][0]);
        acc[0][1] = wmma_bf16(a0, b1, acc[0][1]);
        acc[1][1] = wmma_bf16(a1, b1, acc[1][1]);
        acc[0][2] = wmma_bf16(a0, b2, acc[0][2]);
        acc[1][2] = wmma_bf16(a1, b2, acc[1][2]);
        acc[0][3] = wmma_bf16(a0, b3, acc[0][3]);
        acc[1][3] = wmma_bf16(a1, b3, acc[1][3]);
    }

#pragma unroll
    for (int r = 0; r < 2; ++r)
#pragma unroll
        for (int c = 0; c < 4; ++c)
#pragma unroll
            for (int v = 0; v < 8; ++v) {
                size_t row = (size_t)(mT * 32 + r * 16 + v + 8 * hi);
                out[row * CMODEL + nT * 64 + c * 16 + n] = acc[r][c][v];
            }
}

// ---------------------------------------------------------------------------
// Host launcher
// ---------------------------------------------------------------------------
extern "C" void kernel_launch(void* const* d_in, const int* in_sizes, int n_in,
                              void* d_out, int out_size, void* d_ws, size_t ws_size,
                              hipStream_t stream) {
    const float* x     = (const float*)d_in[0];
    const float* w_qkv = (const float*)d_in[1];
    const float* w_out = (const float*)d_in[2];
    float* out = (float*)d_out;

    char* ws = (char*)d_ws;
    unsigned short* xb    = (unsigned short*)(ws + ((size_t)0  << 20)); // 8 MiB
    unsigned short* wqkvb = (unsigned short*)(ws + ((size_t)8  << 20)); // 6 MiB
    unsigned short* woutb = (unsigned short*)(ws + ((size_t)14 << 20)); // 2 MiB
    unsigned short* Qb    = (unsigned short*)(ws + ((size_t)16 << 20)); // 8 MiB
    unsigned short* Kb    = (unsigned short*)(ws + ((size_t)24 << 20)); // 8 MiB
    unsigned short* Vtb   = (unsigned short*)(ws + ((size_t)32 << 20)); // 8 MiB
    unsigned short* attnb = (unsigned short*)(ws + ((size_t)40 << 20)); // 8 MiB

    // convert inputs to bf16
    int nx = TOKENS * CMODEL;                 // 4194304
    int nw = NQKV * CMODEL;                   // 3145728
    int no = CMODEL * CMODEL;                 // 1048576
    cvt_bf16_kernel<<<(nx + 255) / 256, 256, 0, stream>>>(x, xb, nx);
    cvt_bf16_kernel<<<(nw + 255) / 256, 256, 0, stream>>>(w_qkv, wqkvb, nw);
    cvt_bf16_kernel<<<(no + 255) / 256, 256, 0, stream>>>(w_out, woutb, no);

    // QKV projection: (TOKENS/32)*(NQKV/64) = 128*48 waves, 8 waves/block
    qkv_gemm_kernel<<<(128 * 48) / 8, 256, 0, stream>>>(xb, wqkvb, Qb, Kb, Vtb);

    // flash attention: 32 heads * 128 query blocks = 4096 waves
    attn_kernel<<<4096 / 8, 256, 0, stream>>>(Qb, Kb, Vtb, attnb);

    // output projection: (TOKENS/32)*(CMODEL/64) = 128*16 waves
    out_gemm_kernel<<<(128 * 16) / 8, 256, 0, stream>>>(attnb, woutb, out);
}
